// Alternating_25915832664873
// MI455X (gfx1250) — matrix operand order repaired and checked by reading, not compiled, not run on hardware
//
#include <hip/hip_runtime.h>
#include <hip/hip_bf16.h>

// ============================================================================
// MI455X (gfx1250) implementation of the alternating graph network.
// All GEMM work (8 different 3-layer MLPs) runs on v_wmma_f32_16x16x32_f16:
//   - weights pre-converted once to transposed/padded f16 in workspace
//   - each wave32 owns a 16-row tile; activations live as f16 panels in LDS
//   - A-frag built from LDS per the ISA 16-bit 16x32 A layout (loaded once per
//     k-chunk, reused across all output tiles), B-frag read as packed f16
//     dwords from global (L2/WGP$ resident), C/D f32 accumulators.
// Segment sum/mean aggregations use global_atomic_add_f32 (+ counts).
// ============================================================================

typedef __attribute__((ext_vector_type(16))) _Float16 v16h;
typedef __attribute__((ext_vector_type(8)))  float    v8f;

union FragAB { v16h h; unsigned u[8]; };
union FragCD { v8f  f; float    x[8]; };

struct MlpP {
  const _Float16* w0; const float* b0;   // [64][KP0] , [64]
  const _Float16* w1; const float* b1;   // [64][64]  , [64]
  const _Float16* w2; const float* b2;   // [N2T][64] , [N2T]
};

#define WPB 2   // waves per block (blockDim = 64)

// ---- one WMMA layer: act LDS [16][KP] f16 -> out [16][NOUT]
//      kc-outer / nt-inner: A fragment loaded once per k-chunk, reused for
//      all NOUT/16 accumulators.
template<int KP, int NOUT, bool RELU, typename OUT_T>
__device__ inline void layer_wmma(const _Float16* act,
                                  const _Float16* __restrict__ wT,
                                  const float* __restrict__ bias,
                                  OUT_T* out) {
  const int lane = threadIdx.x & 31;
  const int nl = lane & 15;     // A row m / B col n / D col n
  const int hh = lane >> 4;     // K-half selector
  constexpr int NT = NOUT / 16;
  FragCD c[NT];
#pragma unroll
  for (int nt = 0; nt < NT; ++nt)
#pragma unroll
    for (int v = 0; v < 8; ++v) c[nt].x[v] = 0.0f;

#pragma unroll
  for (int kc = 0; kc < KP; kc += 32) {
    FragAB a;
#pragma unroll
    for (int v = 0; v < 8; ++v) {
      const int ka = kc + ((v >> 2) << 4) + (hh << 3) + ((v & 3) << 1);
      a.u[v] = *(const unsigned*)(act + nl * KP + ka);
    }
#pragma unroll
    for (int nt = 0; nt < NT; ++nt) {
      FragAB b;
#pragma unroll
      for (int v = 0; v < 8; ++v) {
        const int kb = kc + ((v >> 2) << 4) + (hh << 3) + ((v & 3) << 1);
        b.u[v] = *(const unsigned*)(wT + (nt * 16 + nl) * KP + kb);
      }
      c[nt].f = __builtin_amdgcn_wmma_f32_16x16x32_f16(false, a.h, false, b.h,
                                                       (short)0, c[nt].f,
                                                       false, false);
    }
  }

#pragma unroll
  for (int nt = 0; nt < NT; ++nt) {
    const float bs = bias[nt * 16 + nl];
#pragma unroll
    for (int v = 0; v < 8; ++v) {
      float r = c[nt].x[v] + bs;
      if (RELU) r = fmaxf(r, 0.0f);
      out[(v + (hh << 3)) * NOUT + nt * 16 + nl] = (OUT_T)r;
    }
  }
}

// ---- generic 3-layer MLP on a 16-row tile (per-wave LDS buffers)
template<int KP0, int N2T, class FILL, class EMIT>
__device__ inline void run_mlp3(const MlpP p, _Float16* act0, _Float16* actA,
                                _Float16* actB, float* stage, bool doFill,
                                FILL fill, EMIT emit) {
  if (doFill) fill(act0);
  __syncthreads();
  layer_wmma<KP0, 64, true, _Float16>(act0, p.w0, p.b0, actA);
  __syncthreads();
  layer_wmma<64, 64, true, _Float16>(actA, p.w1, p.b1, actB);
  __syncthreads();
  layer_wmma<64, N2T, false, float>(actB, p.w2, p.b2, stage);
  __syncthreads();
  emit(stage);
  __syncthreads();
}

// ============================ prep / util kernels ===========================

__global__ void prep_w(const float* __restrict__ W, int K, int KP, int N, int NT,
                       _Float16* __restrict__ out) {
  int i = blockIdx.x * blockDim.x + threadIdx.x;
  if (i >= NT * KP) return;
  int n = i / KP, k = i % KP;
  out[i] = (_Float16)((k < K && n < N) ? W[k * N + n] : 0.0f);
}

__global__ void prep_b(const float* __restrict__ b, int N, int NT,
                       float* __restrict__ out) {
  int i = blockIdx.x * blockDim.x + threadIdx.x;
  if (i >= NT) return;
  out[i] = (i < N) ? b[i] : 0.0f;
}

__global__ void zero_kernel(float* __restrict__ p, int n) {
  int i = blockIdx.x * blockDim.x + threadIdx.x;
  if (i < n) p[i] = 0.0f;
}

// out[r][0:80] = [a(16) | h(32) | sh[batch or r](32)]
__global__ void concat_kernel(const float* __restrict__ a,
                              const float* __restrict__ h,
                              const float* __restrict__ sh,
                              const int* __restrict__ batch,
                              float* __restrict__ out, int R, int useBatch) {
  int i = blockIdx.x * blockDim.x + threadIdx.x;
  if (i >= R * 80) return;
  int r = i / 80, c = i % 80;
  float v;
  if (c < 16)      v = a[r * 16 + c];
  else if (c < 48) v = h[r * 32 + (c - 16)];
  else { int bidx = useBatch ? batch[r] : r; v = sh[bidx * 32 + (c - 48)]; }
  out[i] = v;
}

// ============================== encoder =====================================
// rows of [R,16] -> MLP(16->64->64->32); outHalf selects f16 or f32 output.
__global__ __launch_bounds__(64) void enc_kernel(const float* __restrict__ in,
                                                 int R, MlpP p, void* outp,
                                                 int outHalf) {
  __shared__ _Float16 act0[WPB][16 * 32];
  __shared__ _Float16 actA[WPB][16 * 64];
  __shared__ _Float16 actB[WPB][16 * 64];
  __shared__ float    stg [WPB][16 * 32];
  const int w = threadIdx.x >> 5, lane = threadIdx.x & 31;
  const int tb = (blockIdx.x * WPB + w) * 16;
  run_mlp3<32, 32>(p, act0[w], actA[w], actB[w], stg[w], true,
    [&](_Float16* a0) {
      for (int i = lane; i < 16 * 32; i += 32) {
        int row = i >> 5, col = i & 31;
        int r = min(tb + row, R - 1);
        a0[i] = (_Float16)((col < 16) ? in[r * 16 + col] : 0.0f);
      }
    },
    [&](float* st) {
      for (int i = lane; i < 16 * 32; i += 32) {
        int row = i >> 5, col = i & 31;
        int r = tb + row;
        if (r < R) {
          if (outHalf) ((_Float16*)outp)[r * 32 + col] = (_Float16)st[i];
          else         ((float*)outp)[r * 32 + col] = st[i];
        }
      }
    });
}

// ============================== edge stage ==================================
// edge_in(240) = [xc[d]-xc[s] | e | e_h | uc[bs][48:80] | uc[bs][0:80]], pad 256
__global__ __launch_bounds__(64) void edge_kernel(
    const float* __restrict__ xc, const float* __restrict__ ef,
    const _Float16* __restrict__ eh, const float* __restrict__ uc,
    const int* __restrict__ src, const int* __restrict__ dst,
    const int* __restrict__ batch, MlpP pe, MlpP pa,
    _Float16* __restrict__ ehNew, float* __restrict__ sumMsg,
    float* __restrict__ cntN, int E) {
  __shared__ _Float16 act0[WPB][16 * 256];
  __shared__ _Float16 actA[WPB][16 * 64];
  __shared__ _Float16 actB[WPB][16 * 64];
  __shared__ float stE[WPB][16 * 32];
  __shared__ float stA[WPB][16 * 32];
  __shared__ int   ids[WPB][16 * 3];
  const int w = threadIdx.x >> 5, lane = threadIdx.x & 31;
  const int tb = (blockIdx.x * WPB + w) * 16;
  if (lane < 16) {
    int e = min(tb + lane, E - 1);
    int s = src[e], d = dst[e];
    ids[w][lane * 3 + 0] = s;
    ids[w][lane * 3 + 1] = d;
    ids[w][lane * 3 + 2] = batch[s];
  }
  __syncthreads();
  auto fill = [&](_Float16* a0) {
    for (int i = lane; i < 16 * 256; i += 32) {
      int row = i >> 8, col = i & 255;
      int s = ids[w][row * 3 + 0], d = ids[w][row * 3 + 1],
          bs = ids[w][row * 3 + 2];
      int e = min(tb + row, E - 1);
      float v;
      if (col < 80)       v = xc[d * 80 + col] - xc[s * 80 + col];
      else if (col < 96)  v = ef[e * 16 + (col - 80)];
      else if (col < 128) v = (float)eh[e * 32 + (col - 96)];
      else if (col < 160) v = uc[bs * 80 + 48 + (col - 128)];
      else if (col < 240) v = uc[bs * 80 + (col - 160)];
      else                v = 0.0f;
      a0[i] = (_Float16)v;
    }
  };
  // rec_edge -> e_h (kept in stE + written out as f16)
  run_mlp3<256, 32>(pe, act0[w], actA[w], actB[w], stE[w], true, fill,
    [&](float* st) {
      for (int i = lane; i < 16 * 32; i += 32) {
        int row = i >> 5, col = i & 31;
        int e = tb + row;
        if (e < E) ehNew[e * 32 + col] = (_Float16)st[i];
      }
    });
  // rec_att -> gate; msg = e_h * sigmoid(att); segment-sum via f32 atomics
  run_mlp3<256, 32>(pa, act0[w], actA[w], actB[w], stA[w], false, fill,
    [&](float* st) {
      for (int i = lane; i < 16 * 32; i += 32) {
        int row = i >> 5, col = i & 31;
        int e = tb + row;
        float g = 1.0f / (1.0f + __expf(-st[i]));
        float m = stE[w][i] * g;
        if (e < E) atomicAdd(&sumMsg[ids[w][row * 3 + 1] * 32 + col], m);
      }
      if (lane < 16) {
        int e = tb + lane;
        if (e < E) atomicAdd(&cntN[ids[w][lane * 3 + 1]], 1.0f);
      }
    });
}

// ============================== node stage ==================================
// node_in(224) = [sum(32) | mean(32) | xc(80) | uc[batch](80)]
__global__ __launch_bounds__(64) void node_kernel(
    const float* __restrict__ xc, const float* __restrict__ uc,
    const float* __restrict__ sumMsg, const float* __restrict__ cntN,
    const int* __restrict__ batch, MlpP p, float* __restrict__ xhNew,
    float* __restrict__ sumXh, float* __restrict__ cntB, int N) {
  __shared__ _Float16 act0[WPB][16 * 224];
  __shared__ _Float16 actA[WPB][16 * 64];
  __shared__ _Float16 actB[WPB][16 * 64];
  __shared__ float stg[WPB][16 * 32];
  __shared__ int   bid[WPB][16];
  __shared__ float rc [WPB][16];
  const int w = threadIdx.x >> 5, lane = threadIdx.x & 31;
  const int tb = (blockIdx.x * WPB + w) * 16;
  if (lane < 16) {
    int n = min(tb + lane, N - 1);
    bid[w][lane] = batch[n];
    rc[w][lane] = 1.0f / fmaxf(cntN[n], 1.0f);
  }
  __syncthreads();
  run_mlp3<224, 32>(p, act0[w], actA[w], actB[w], stg[w], true,
    [&](_Float16* a0) {
      for (int i = lane; i < 16 * 224; i += 32) {
        int row = i / 224, col = i % 224;
        int n = min(tb + row, N - 1);
        float v;
        if (col < 32)       v = sumMsg[n * 32 + col];
        else if (col < 64)  v = sumMsg[n * 32 + (col - 32)] * rc[w][row];
        else if (col < 144) v = xc[n * 80 + (col - 64)];
        else                v = uc[bid[w][row] * 80 + (col - 144)];
        a0[i] = (_Float16)v;
      }
    },
    [&](float* st) {
      for (int i = lane; i < 16 * 32; i += 32) {
        int row = i >> 5, col = i & 31;
        int n = tb + row;
        if (n < N) {
          xhNew[n * 32 + col] = st[i];
          atomicAdd(&sumXh[bid[w][row] * 32 + col], st[i]);
        }
      }
      if (lane < 16) {
        int n = tb + lane;
        if (n < N) atomicAdd(&cntB[bid[w][lane]], 1.0f);
      }
    });
}

// ============================== global stage ================================
// glob_in(144,pad160) = [sum(32) | mean(32) | uc(80)]
__global__ __launch_bounds__(64) void glob_kernel(
    const float* __restrict__ uc, const float* __restrict__ sumXh,
    const float* __restrict__ cntB, MlpP p, float* __restrict__ uhNew, int Bn) {
  __shared__ _Float16 act0[WPB][16 * 160];
  __shared__ _Float16 actA[WPB][16 * 64];
  __shared__ _Float16 actB[WPB][16 * 64];
  __shared__ float stg[WPB][16 * 32];
  __shared__ float rc [WPB][16];
  const int w = threadIdx.x >> 5, lane = threadIdx.x & 31;
  const int tb = (blockIdx.x * WPB + w) * 16;
  if (lane < 16) {
    int b = min(tb + lane, Bn - 1);
    rc[w][lane] = 1.0f / fmaxf(cntB[b], 1.0f);
  }
  __syncthreads();
  run_mlp3<160, 32>(p, act0[w], actA[w], actB[w], stg[w], true,
    [&](_Float16* a0) {
      for (int i = lane; i < 16 * 160; i += 32) {
        int row = i / 160, col = i % 160;
        int b = min(tb + row, Bn - 1);
        float v;
        if (col < 32)       v = sumXh[b * 32 + col];
        else if (col < 64)  v = sumXh[b * 32 + (col - 32)] * rc[w][row];
        else if (col < 144) v = uc[b * 80 + (col - 64)];
        else                v = 0.0f;
        a0[i] = (_Float16)v;
      }
    },
    [&](float* st) {
      for (int i = lane; i < 16 * 32; i += 32) {
        int row = i >> 5, col = i & 31;
        int b = tb + row;
        if (b < Bn) uhNew[b * 32 + col] = st[i];
      }
    });
}

// ============================== decoder =====================================
__global__ __launch_bounds__(64) void dec_kernel(const float* __restrict__ uh,
                                                 MlpP p, float* __restrict__ out,
                                                 int Bn) {
  __shared__ _Float16 act0[WPB][16 * 32];
  __shared__ _Float16 actA[WPB][16 * 64];
  __shared__ _Float16 actB[WPB][16 * 64];
  __shared__ float stg[WPB][16 * 16];
  const int w = threadIdx.x >> 5, lane = threadIdx.x & 31;
  const int tb = (blockIdx.x * WPB + w) * 16;
  run_mlp3<32, 16>(p, act0[w], actA[w], actB[w], stg[w], true,
    [&](_Float16* a0) {
      for (int i = lane; i < 16 * 32; i += 32) {
        int row = i >> 5, col = i & 31;
        int b = min(tb + row, Bn - 1);
        a0[i] = (_Float16)uh[b * 32 + col];
      }
    },
    [&](float* st) {
      for (int i = lane; i < 16 * 16; i += 32) {
        int row = i >> 4, col = i & 15;
        int b = tb + row;
        if (b < Bn && col < 2) out[b * 2 + col] = st[i];
      }
    });
}

// ============================== host driver =================================

static inline int cdiv(int a, int b) { return (a + b - 1) / b; }

extern "C" void kernel_launch(void* const* d_in, const int* in_sizes, int n_in,
                              void* d_out, int out_size, void* d_ws, size_t ws_size,
                              hipStream_t stream) {
  (void)n_in; (void)out_size; (void)ws_size;
  const int N  = in_sizes[0] / 16;   // 30000
  const int E  = in_sizes[2] / 16;   // 300000
  const int Bn = in_sizes[3] / 16;   // 256

  const float* x1  = (const float*)d_in[0];
  const int*   ei1 = (const int*)  d_in[1];
  const float* e1  = (const float*)d_in[2];
  const float* u1  = (const float*)d_in[3];
  const int*   b1  = (const int*)  d_in[4];
  const float* x2  = (const float*)d_in[5];
  const int*   ei2 = (const int*)  d_in[6];
  const float* e2  = (const float*)d_in[7];
  const float* u2  = (const float*)d_in[8];
  const int*   b2  = (const int*)  d_in[9];
  // params in setup_inputs() insertion order:
  //   enc_x, enc_e, enc_u, rec_edge, rec_att, rec_node, rec_glob, dec
  // each as W0,b0,W1,b1,W2,b2
  const float* P[48];
  for (int i = 0; i < 48; ++i) P[i] = (const float*)d_in[10 + i];

  const int KIN[8]   = {16, 16, 16, 240, 240, 224, 144, 32};
  const int KP[8]    = {32, 32, 32, 256, 256, 224, 160, 32};
  const int NOUT2[8] = {32, 32, 32, 32, 32, 32, 32, 2};
  const int N2T[8]   = {32, 32, 32, 32, 32, 32, 32, 16};

  // ---- workspace layout (bump allocator) ----
  char* wp = (char*)d_ws;
  auto alloc = [&](size_t bytes) {
    void* r = (void*)wp;
    wp += (bytes + 255) & ~(size_t)255;
    return r;
  };
  float* x1h[2] = {(float*)alloc((size_t)N * 32 * 4), (float*)alloc((size_t)N * 32 * 4)};
  float* x2h[2] = {(float*)alloc((size_t)N * 32 * 4), (float*)alloc((size_t)N * 32 * 4)};
  _Float16* e1h[2] = {(_Float16*)alloc((size_t)E * 32 * 2), (_Float16*)alloc((size_t)E * 32 * 2)};
  _Float16* e2h[2] = {(_Float16*)alloc((size_t)E * 32 * 2), (_Float16*)alloc((size_t)E * 32 * 2)};
  float* u1h[2] = {(float*)alloc((size_t)Bn * 32 * 4), (float*)alloc((size_t)Bn * 32 * 4)};
  float* u2h[2] = {(float*)alloc((size_t)Bn * 32 * 4), (float*)alloc((size_t)Bn * 32 * 4)};
  float* xcb    = (float*)alloc((size_t)N * 80 * 4);
  float* ucb    = (float*)alloc((size_t)Bn * 80 * 4);
  float* sumMsg = (float*)alloc((size_t)N * 32 * 4);
  float* cntN   = (float*)alloc((size_t)N * 4);
  float* sumXh  = (float*)alloc((size_t)Bn * 32 * 4);
  float* cntB   = (float*)alloc((size_t)Bn * 4);

  // ---- one-time weight conversion: transposed, K-padded f16 ----
  MlpP mp[8];
  for (int m = 0; m < 8; ++m) {
    _Float16* w0 = (_Float16*)alloc((size_t)64 * KP[m] * 2);
    _Float16* w1 = (_Float16*)alloc((size_t)64 * 64 * 2);
    _Float16* w2 = (_Float16*)alloc((size_t)N2T[m] * 64 * 2);
    float* c0 = (float*)alloc(64 * 4);
    float* c1 = (float*)alloc(64 * 4);
    float* c2 = (float*)alloc((size_t)N2T[m] * 4);
    prep_w<<<cdiv(64 * KP[m], 256), 256, 0, stream>>>(P[m * 6 + 0], KIN[m], KP[m], 64, 64, w0);
    prep_w<<<cdiv(64 * 64, 256), 256, 0, stream>>>(P[m * 6 + 2], 64, 64, 64, 64, w1);
    prep_w<<<cdiv(N2T[m] * 64, 256), 256, 0, stream>>>(P[m * 6 + 4], 64, 64, NOUT2[m], N2T[m], w2);
    prep_b<<<1, 64, 0, stream>>>(P[m * 6 + 1], 64, 64, c0);
    prep_b<<<1, 64, 0, stream>>>(P[m * 6 + 3], 64, 64, c1);
    prep_b<<<1, 64, 0, stream>>>(P[m * 6 + 5], NOUT2[m], N2T[m], c2);
    mp[m].w0 = w0; mp[m].b0 = c0;
    mp[m].w1 = w1; mp[m].b1 = c1;
    mp[m].w2 = w2; mp[m].b2 = c2;
  }

  const int tilesN = cdiv(N, 16), tilesE = cdiv(E, 16), tilesB = cdiv(Bn, 16);

  // ---- encoders ----
  enc_kernel<<<cdiv(tilesN, WPB), 64, 0, stream>>>(x1, N, mp[0], (void*)x1h[0], 0);
  enc_kernel<<<cdiv(tilesE, WPB), 64, 0, stream>>>(e1, E, mp[1], (void*)e1h[0], 1);
  enc_kernel<<<cdiv(tilesB, WPB), 64, 0, stream>>>(u1, Bn, mp[2], (void*)u1h[0], 0);
  enc_kernel<<<cdiv(tilesN, WPB), 64, 0, stream>>>(x2, N, mp[0], (void*)x2h[0], 0);
  enc_kernel<<<cdiv(tilesE, WPB), 64, 0, stream>>>(e2, E, mp[1], (void*)e2h[0], 1);
  enc_kernel<<<cdiv(tilesB, WPB), 64, 0, stream>>>(u2, Bn, mp[2], (void*)u2h[0], 0);

  int ix1 = 0, ie1 = 0, iu1 = 0, ix2 = 0, ie2 = 0, iu2 = 0;

  auto run_pass = [&](const float* x, const float* ef, const float* u,
                      const int* srcp, const int* dstp, const int* bat,
                      float** xh, _Float16** eh, float** uh,
                      int& ix, int& ie, int& iu, const float* shared) {
    concat_kernel<<<cdiv(N * 80, 256), 256, 0, stream>>>(x, xh[ix], shared, bat, xcb, N, 1);
    concat_kernel<<<cdiv(Bn * 80, 256), 256, 0, stream>>>(u, uh[iu], shared, (const int*)nullptr, ucb, Bn, 0);
    zero_kernel<<<cdiv(N * 32, 256), 256, 0, stream>>>(sumMsg, N * 32);
    zero_kernel<<<cdiv(N, 256), 256, 0, stream>>>(cntN, N);
    edge_kernel<<<cdiv(tilesE, WPB), 64, 0, stream>>>(xcb, ef, eh[ie], ucb,
        srcp, dstp, bat, mp[3], mp[4], eh[1 - ie], sumMsg, cntN, E);
    zero_kernel<<<cdiv(Bn * 32, 256), 256, 0, stream>>>(sumXh, Bn * 32);
    zero_kernel<<<cdiv(Bn, 256), 256, 0, stream>>>(cntB, Bn);
    node_kernel<<<cdiv(tilesN, WPB), 64, 0, stream>>>(xcb, ucb, sumMsg, cntN,
        bat, mp[5], xh[1 - ix], sumXh, cntB, N);
    glob_kernel<<<cdiv(tilesB, WPB), 64, 0, stream>>>(ucb, sumXh, cntB, mp[6],
        uh[1 - iu], Bn);
    ix = 1 - ix; ie = 1 - ie; iu = 1 - iu;
  };

  // graph 1 (shared = encoder u2h), two passes
  for (int p = 0; p < 2; ++p)
    run_pass(x1, e1, u1, ei1, ei1 + E, b1, x1h, e1h, u1h, ix1, ie1, iu1, u2h[iu2]);
  // graph 2 (shared = updated u1h), two passes
  for (int p = 0; p < 2; ++p)
    run_pass(x2, e2, u2, ei2, ei2 + E, b2, x2h, e2h, u2h, ix2, ie2, iu2, u1h[iu1]);

  // decoder on final u2h -> [1, B, 2]
  dec_kernel<<<cdiv(tilesB, WPB), 64, 0, stream>>>(u2h[iu2], mp[7], (float*)d_out, Bn);
}
